// Decoder_71004399337777
// MI455X (gfx1250) — matrix operand order repaired and checked by reading, not compiled, not run on hardware
//
#include <hip/hip_runtime.h>

typedef __bf16 bf16_t;
typedef __attribute__((ext_vector_type(16))) __bf16 v16bf;
typedef __attribute__((ext_vector_type(8)))  __bf16 v8bf;
typedef __attribute__((ext_vector_type(8)))  float  v8f;

#define B_SZ 64
#define T_SZ 32
#define E_SZ 512
#define H_SZ 1024
#define V_SZ 10000
#define F_SZ 2048

#if defined(__has_builtin)
#if __has_builtin(__builtin_amdgcn_tanhf)
#define FAST_TANH(x) __builtin_amdgcn_tanhf(x)
#endif
#endif
#ifndef FAST_TANH
#define FAST_TANH(x) tanhf(x)
#endif

__device__ __forceinline__ float sigmoidf_(float x) { return 1.0f / (1.0f + __expf(-x)); }

// A-fragment: 16x32 bf16, M rows = lane&15; lanes 0-15 hold K=k0+[0..15];
// lanes 16-31 hold K=k0+[16..31].
__device__ __forceinline__ v16bf load_a_frag(const bf16_t* __restrict__ X, int ldx,
                                             int m0, int k0, int lane) {
    int row = m0 + (lane & 15);
    int kb  = k0 + ((lane >> 4) << 4);
    const bf16_t* p = X + (size_t)row * ldx + kb;
    v8bf lo = *(const v8bf*)(p);
    v8bf hi = *(const v8bf*)(p + 8);
    v16bf a;
#pragma unroll
    for (int i = 0; i < 8; ++i) { a[i] = lo[i]; a[8 + i] = hi[i]; }
    return a;
}

// B-fragment: 32x16 (KxN) from row-major W[N,K]; lanes 0-15: N=lane, K=k0..k0+15;
// lanes 16-31: N=lane-16, K=k0+16..k0+31. One contiguous 32-byte load per lane.
__device__ __forceinline__ v16bf load_b_frag(const bf16_t* __restrict__ W, int ldw,
                                             int n0, int k0, int lane) {
    int col = n0 + (lane & 15);
    int kb  = k0 + ((lane >> 4) << 4);
    return *(const v16bf*)(W + (size_t)col * ldw + kb);
}

// Same but with the row index clamped in-bounds (for the ragged N=10000 edge).
__device__ __forceinline__ v16bf load_b_frag_clamp(const bf16_t* __restrict__ W, int ldw,
                                                   int n0, int k0, int lane, int nmax) {
    int col = n0 + (lane & 15);
    if (col > nmax) col = nmax;
    int kb  = k0 + ((lane >> 4) << 4);
    return *(const v16bf*)(W + (size_t)col * ldw + kb);
}

__global__ void convert_bf16_kernel(const float* __restrict__ src, bf16_t* __restrict__ dst, int n) {
    int i = blockIdx.x * blockDim.x + threadIdx.x;
    int stride = gridDim.x * blockDim.x;
    for (; i < n; i += stride) dst[i] = (bf16_t)src[i];
}

// emb[t][b][e] = table[captions[b][t]][e], converted to bf16
__global__ void embed_kernel(const int* __restrict__ captions, const float* __restrict__ table,
                             bf16_t* __restrict__ emb) {
    int i = blockIdx.x * blockDim.x + threadIdx.x;
    int stride = gridDim.x * blockDim.x;
    const int total = T_SZ * B_SZ * E_SZ;
    for (; i < total; i += stride) {
        int e  = i & (E_SZ - 1);
        int tb = i >> 9;           // t*64 + b
        int b  = tb & (B_SZ - 1);
        int t  = tb >> 6;
        int tok = captions[b * T_SZ + t];
        emb[i] = (bf16_t)table[(size_t)tok * E_SZ + e];
    }
}

// D[64, 1024] = X[64, 2048] @ W[1024, 2048]^T + bias ; write fp32 and/or bf16
__global__ __launch_bounds__(256) void init_gemm_kernel(
    const bf16_t* __restrict__ X, const bf16_t* __restrict__ W,
    const float* __restrict__ bias, float* __restrict__ out_f32, bf16_t* __restrict__ out_bf16) {
    int lane = threadIdx.x & 31;
    int wid  = threadIdx.x >> 5;
    int m0   = (wid & 3) * 16;
    int ncol = blockIdx.x * 32 + (wid >> 2) * 16;
    v8f acc = {0.f, 0.f, 0.f, 0.f, 0.f, 0.f, 0.f, 0.f};
    for (int k0 = 0; k0 < F_SZ; k0 += 32) {
        v16bf a = load_a_frag(X, F_SZ, m0, k0, lane);
        v16bf b = load_b_frag(W, F_SZ, ncol, k0, lane);
        acc = __builtin_amdgcn_wmma_f32_16x16x32_bf16(false, a, false, b, (short)0, acc, false, false);
    }
    int col = ncol + (lane & 15);
    float bv = bias[col];
    int rbase = m0 + ((lane >> 4) << 3);
#pragma unroll
    for (int i = 0; i < 8; ++i) {
        float v = acc[i] + bv;
        int idx = (rbase + i) * H_SZ + col;
        if (out_f32)  out_f32[idx]  = v;
        if (out_bf16) out_bf16[idx] = (bf16_t)v;
    }
}

// Fused gates GEMM + LSTM cell. Block owns hidden columns [blockIdx.x*32, +32).
// Each wave: one 16x16 (m,n) subtile, 4 accumulators (i,f,g,o) sharing the A-fragment.
__global__ __launch_bounds__(256) void gates_cell_kernel(
    const bf16_t* __restrict__ x,    // [64, 512]
    const bf16_t* __restrict__ hin,  // [64, 1024]
    const bf16_t* __restrict__ Wih,  // [4096, 512]
    const bf16_t* __restrict__ Whh,  // [4096, 1024]
    const float* __restrict__ b_ih, const float* __restrict__ b_hh,
    float* __restrict__ c,           // [64, 1024] in/out (block-private columns)
    bf16_t* __restrict__ hout) {     // [64, 1024]
    int lane = threadIdx.x & 31;
    int wid  = threadIdx.x >> 5;
    int m0   = (wid & 3) * 16;
    int ncol = blockIdx.x * 32 + (wid >> 2) * 16;

    v8f acc[4];
#pragma unroll
    for (int g = 0; g < 4; ++g) acc[g] = (v8f){0.f, 0.f, 0.f, 0.f, 0.f, 0.f, 0.f, 0.f};

    for (int k0 = 0; k0 < E_SZ; k0 += 32) {
        v16bf a = load_a_frag(x, E_SZ, m0, k0, lane);
#pragma unroll
        for (int g = 0; g < 4; ++g) {
            v16bf b = load_b_frag(Wih, E_SZ, g * H_SZ + ncol, k0, lane);
            acc[g] = __builtin_amdgcn_wmma_f32_16x16x32_bf16(false, a, false, b, (short)0, acc[g], false, false);
        }
    }
    for (int k0 = 0; k0 < H_SZ; k0 += 32) {
        v16bf a = load_a_frag(hin, H_SZ, m0, k0, lane);
#pragma unroll
        for (int g = 0; g < 4; ++g) {
            v16bf b = load_b_frag(Whh, H_SZ, g * H_SZ + ncol, k0, lane);
            acc[g] = __builtin_amdgcn_wmma_f32_16x16x32_bf16(false, a, false, b, (short)0, acc[g], false, false);
        }
    }

    int col = ncol + (lane & 15);
    float bi = b_ih[col]            + b_hh[col];
    float bf = b_ih[H_SZ + col]     + b_hh[H_SZ + col];
    float bg = b_ih[2 * H_SZ + col] + b_hh[2 * H_SZ + col];
    float bo = b_ih[3 * H_SZ + col] + b_hh[3 * H_SZ + col];
    int rbase = m0 + ((lane >> 4) << 3);
#pragma unroll
    for (int i = 0; i < 8; ++i) {
        int idx = (rbase + i) * H_SZ + col;
        float iv = sigmoidf_(acc[0][i] + bi);
        float fv = sigmoidf_(acc[1][i] + bf);
        float gv = FAST_TANH(acc[2][i] + bg);
        float ov = sigmoidf_(acc[3][i] + bo);
        float cc = fv * c[idx] + iv * gv;
        c[idx] = cc;
        hout[idx] = (bf16_t)(ov * FAST_TANH(cc));
    }
}

// out[64, 10000] = sigmoid(h[64,1024] @ Wfc[10000,1024]^T + b_fc)
// Block covers 64x128 of C; each wave owns 16x64 (4 N-tiles) sharing one A-fragment
// per K-step -> 4 WMMAs per A-load (same intensity as the gates kernel).
__global__ __launch_bounds__(256) void logits_kernel(
    const bf16_t* __restrict__ h, const bf16_t* __restrict__ Wfc,
    const float* __restrict__ b_fc, float* __restrict__ out) {
    int lane = threadIdx.x & 31;
    int wid  = threadIdx.x >> 5;
    int m0   = (wid & 3) * 16;
    int nbase = blockIdx.x * 128 + (wid >> 2) * 64;   // wave's first N-tile

    v8f acc[4];
#pragma unroll
    for (int j = 0; j < 4; ++j) acc[j] = (v8f){0.f, 0.f, 0.f, 0.f, 0.f, 0.f, 0.f, 0.f};

    for (int k0 = 0; k0 < H_SZ; k0 += 32) {
        v16bf a = load_a_frag(h, H_SZ, m0, k0, lane);
#pragma unroll
        for (int j = 0; j < 4; ++j) {
            v16bf b = load_b_frag_clamp(Wfc, H_SZ, nbase + j * 16, k0, lane, V_SZ - 1);
            acc[j] = __builtin_amdgcn_wmma_f32_16x16x32_bf16(false, a, false, b, (short)0, acc[j], false, false);
        }
    }

    int rbase = m0 + ((lane >> 4) << 3);
#pragma unroll
    for (int j = 0; j < 4; ++j) {
        int col = nbase + j * 16 + (lane & 15);
        if (col < V_SZ) {
            float bv = b_fc[col];
#pragma unroll
            for (int i = 0; i < 8; ++i)
                out[(size_t)(rbase + i) * V_SZ + col] = sigmoidf_(acc[j][i] + bv);
        }
    }
}

extern "C" void kernel_launch(void* const* d_in, const int* in_sizes, int n_in,
                              void* d_out, int out_size, void* d_ws, size_t ws_size,
                              hipStream_t stream) {
    (void)in_sizes; (void)n_in; (void)out_size; (void)ws_size;
    const float* features    = (const float*)d_in[0];
    const int*   captions    = (const int*)  d_in[1];
    const float* embed_table = (const float*)d_in[2];
    const float* W_init_h    = (const float*)d_in[3];
    const float* b_init_h    = (const float*)d_in[4];
    const float* W_init_c    = (const float*)d_in[5];
    const float* b_init_c    = (const float*)d_in[6];
    const float* W_ih        = (const float*)d_in[7];
    const float* b_ih        = (const float*)d_in[8];
    const float* W_hh        = (const float*)d_in[9];
    const float* b_hh        = (const float*)d_in[10];
    const float* W_fc        = (const float*)d_in[11];
    const float* b_fc        = (const float*)d_in[12];
    float* out = (float*)d_out;

    char* ws = (char*)d_ws;
    auto alloc = [&](size_t bytes) -> char* {
        char* p = ws; ws += (bytes + 255) & ~(size_t)255; return p;
    };
    bf16_t* Wih_b   = (bf16_t*)alloc((size_t)4 * H_SZ * E_SZ * 2);
    bf16_t* Whh_b   = (bf16_t*)alloc((size_t)4 * H_SZ * H_SZ * 2);
    bf16_t* Wfc_b   = (bf16_t*)alloc((size_t)V_SZ * H_SZ * 2);
    bf16_t* WinH_b  = (bf16_t*)alloc((size_t)H_SZ * F_SZ * 2);
    bf16_t* WinC_b  = (bf16_t*)alloc((size_t)H_SZ * F_SZ * 2);
    bf16_t* feat_b  = (bf16_t*)alloc((size_t)B_SZ * F_SZ * 2);
    bf16_t* emb_b   = (bf16_t*)alloc((size_t)T_SZ * B_SZ * E_SZ * 2);
    bf16_t* hbuf0   = (bf16_t*)alloc((size_t)B_SZ * H_SZ * 2);
    bf16_t* hbuf1   = (bf16_t*)alloc((size_t)B_SZ * H_SZ * 2);
    float*  cbuf    = (float*) alloc((size_t)B_SZ * H_SZ * 4);
    bf16_t* hbuf[2] = {hbuf0, hbuf1};

    auto conv = [&](const float* s, bf16_t* d, int n) {
        int grid = (n + 255) / 256; if (grid > 8192) grid = 8192;
        convert_bf16_kernel<<<grid, 256, 0, stream>>>(s, d, n);
    };
    conv(W_ih,     Wih_b,  4 * H_SZ * E_SZ);
    conv(W_hh,     Whh_b,  4 * H_SZ * H_SZ);
    conv(W_fc,     Wfc_b,  V_SZ * H_SZ);
    conv(W_init_h, WinH_b, H_SZ * F_SZ);
    conv(W_init_c, WinC_b, H_SZ * F_SZ);
    conv(features, feat_b, B_SZ * F_SZ);

    embed_kernel<<<2048, 256, 0, stream>>>(captions, embed_table, emb_b);

    // h0 (bf16 only), c0 (fp32 only)
    init_gemm_kernel<<<H_SZ / 32, 256, 0, stream>>>(feat_b, WinH_b, b_init_h, nullptr, hbuf[0]);
    init_gemm_kernel<<<H_SZ / 32, 256, 0, stream>>>(feat_b, WinC_b, b_init_c, cbuf, nullptr);

    for (int t = 0; t < T_SZ; ++t) {
        const bf16_t* x_t  = emb_b + (size_t)t * B_SZ * E_SZ;
        bf16_t* hin  = hbuf[t & 1];
        bf16_t* hout = hbuf[(t + 1) & 1];
        gates_cell_kernel<<<H_SZ / 32, 256, 0, stream>>>(
            x_t, hin, Wih_b, Whh_b, b_ih, b_hh, cbuf, hout);
        logits_kernel<<<(V_SZ + 127) / 128, 256, 0, stream>>>(
            hout, Wfc_b, b_fc, out + (size_t)t * B_SZ * V_SZ);
    }
}
